// SemanticMapSelfLSTMModel_16028817949068
// MI455X (gfx1250) — matrix operand order repaired and checked by reading, not compiled, not run on hardware
//
#include <hip/hip_runtime.h>

typedef __attribute__((ext_vector_type(16))) _Float16 v16h;
typedef __attribute__((ext_vector_type(8)))  _Float16 v8h;
typedef __attribute__((ext_vector_type(8)))  float    v8f;

#define OBS_LEN  20
#define PRED_LEN 30
#define EMBED    64
#define HIDDEN   128
#define GATES    512      // 4*HIDDEN
#define CNN_OUT  2048
#define KTOT     192      // EMBED + HIDDEN
#define MROWS    16       // rows per block (one 16-row WMMA tile)
#define NTHREADS 128      // 4 wave32 -> 1 wave per SIMD32, ~1024 VGPRs each

// LDS strides padded for bank-conflict-free ds_read_b128 patterns
#define WB_STRIDE    520   // f16 row stride for K-major weights (512 + 8)
#define ACOMB_STRIDE 200   // f16 row stride for [x_emb|h] (192 + 8)
#define HF_STRIDE    132   // f32 row stride for f32 h copy (128 + 4)

#define WB_OFF    0                       // 192*520*2 = 199680
#define ACOMB_OFF 199680                  // 16*200*2  = 6400
#define HF_OFF    206080                  // 16*132*4  = 8448
#define PWH_OFF   214528                  // 2*128*4   = 1024
#define EMBW_OFF  215552                  // 64*2*4    = 512
#define EMBB_OFF  216064                  // 64*4      = 256
#define BASED_OFF 216320                  // 16*2*4    = 128
#define DISP_OFF  216448                  // 16*2*4    = 128
#define POS_OFF   216576                  // 16*2*4    = 128
#define RED_OFF   216704                  // 16*8*2*4  = 1024
#define SMEM_BYTES 217728                 // ~212.6 KB of the 320 KB WGP LDS

__device__ __forceinline__ float sigf(float x){ return 1.0f/(1.0f + __expf(-x)); }
__device__ __forceinline__ float tanh_fast(float x){
  x = fminf(fmaxf(x, -15.0f), 15.0f);
  float e = __expf(2.0f*x);
  return (e - 1.0f)/(e + 1.0f);
}

// A fragment, 16x32 f16 (ISA 7.12.2): lanes 0-15 row M=lane, K = k0+{0..7,16..23};
// lanes 16-31 same rows, K = k0+{8..15,24..31}. Two 16B ds reads per lane.
__device__ __forceinline__ v16h load_frag_A(const _Float16* Acomb, int k0, int lane){
  const int lrow = lane & 15, half = lane >> 4;
  const _Float16* p = Acomb + lrow*ACOMB_STRIDE + k0 + half*8;
  v8h lo = *(const v8h*)p;
  v8h hi = *(const v8h*)(p + 16);
  v16h r;
  #pragma unroll
  for (int i=0;i<8;++i){ r[i]=lo[i]; r[8+i]=hi[i]; }
  return r;
}

// B fragment, 32x16 f16: lane holds column N = cb+(lane&15), K = k0+{0..7,16..23}
// (lanes 0-15) or k0+{8..15,24..31} (lanes 16-31), matching the A fragment.
__device__ __forceinline__ v16h load_frag_B(const _Float16* Wb, int k0, int cb, int lane){
  const _Float16* p = Wb + (k0 + (lane >> 4)*8)*WB_STRIDE + cb + (lane & 15);
  v16h r;
  #pragma unroll
  for (int i=0;i<8;++i){ r[i]=p[i*WB_STRIDE]; r[8+i]=p[(16+i)*WB_STRIDE]; }
  return r;
}

// One LSTM step: this wave's 16 rows x 32 hidden cols, all four gates.
// Weight (B) fragments are fully VGPR-resident; the whole A matrix (6 fragments,
// 12 ds_read_b128) is issued up front so the 48 back-to-back v_wmma drain the
// loads with partial dscnt waits instead of stalling per chunk (we run 1 wave
// per SIMD, so LDS latency must be hidden inside the instruction stream).
// C-tile layout (ISA 7.12.2): VGPR v, lane l -> row = v + 8*(l>>4), col = l&15,
// so i/f/g/o of one element share a lane/VGPR slot -> pure-register nonlinearity.
template<bool MASKED>
__device__ __forceinline__ void lstm_step(const _Float16* __restrict__ Acomb,
                                          const v16h (&Ball)[6][8],
                                          const float bias[8],
                                          float h_reg[2][8], float c_reg[2][8],
                                          const int hist[8], int thresh, int lane)
{
  v16h afr[6];
  #pragma unroll
  for (int kc=0; kc<6; ++kc)
    afr[kc] = load_frag_A(Acomb, kc*32, lane);      // all A loads in flight

  v8f acc[8];
  #pragma unroll
  for (int tt=0; tt<8; ++tt){
    v8f c;
    #pragma unroll
    for (int i=0;i<8;++i) c[i] = bias[tt];          // fold b_ih + b_hh
    acc[tt] = c;
  }
  #pragma unroll
  for (int kc=0; kc<6; ++kc){                       // K = 192 = 6 x 32
    #pragma unroll
    for (int tt=0; tt<8; ++tt){
      acc[tt] = __builtin_amdgcn_wmma_f32_16x16x32_f16(
          false, afr[kc], false, Ball[kc][tt], (short)0, acc[tt], false, false);
    }
  }
  #pragma unroll
  for (int s=0;s<2;++s){
    #pragma unroll
    for (int v=0; v<8; ++v){
      float gi = acc[0+s][v], gf = acc[2+s][v], gg = acc[4+s][v], go = acc[6+s][v];
      float c2 = sigf(gf)*c_reg[s][v] + sigf(gi)*tanh_fast(gg);
      float h2 = sigf(go)*tanh_fast(c2);
      if (MASKED){
        bool m = hist[v] > thresh;
        c_reg[s][v] = m ? c2 : c_reg[s][v];
        h_reg[s][v] = m ? h2 : h_reg[s][v];
      } else {
        c_reg[s][v] = c2;
        h_reg[s][v] = h2;
      }
    }
  }
}

__device__ __forceinline__ void store_h(_Float16* Acomb, float* hf,
                                        const float h_reg[2][8],
                                        int cg, int lrow, int half){
  #pragma unroll
  for (int s=0;s<2;++s){
    int hcol = cg*32 + s*16 + lrow;
    #pragma unroll
    for (int v=0; v<8; ++v){
      int row = v + 8*half;
      Acomb[row*ACOMB_STRIDE + EMBED + hcol] = (_Float16)h_reg[s][v];
      hf[row*HF_STRIDE + hcol] = h_reg[s][v];
    }
  }
}

// x_emb = relu(rel @ embed_W^T + embed_b): 8 columns per thread, one ds_store_b128.
__device__ __forceinline__ void write_xemb(_Float16* Acomb, const float* embW,
                                           const float* embB,
                                           int row, int c0, float rx, float ry){
  v8h xe;
  #pragma unroll
  for (int i=0;i<8;++i){
    int e = c0 + i;
    float x = fmaf(rx, embW[e*2+0], fmaf(ry, embW[e*2+1], embB[e]));
    xe[i] = (_Float16)fmaxf(x, 0.0f);
  }
  *(v8h*)(Acomb + row*ACOMB_STRIDE + c0) = xe;
}

extern "C" __global__ void __launch_bounds__(NTHREADS)
social_lstm_wmma(const float* __restrict__ img,
                 const float* __restrict__ obs_pos,
                 const int*   __restrict__ obs_hist,
                 const float* __restrict__ obs_rel,
                 const float* __restrict__ h0,
                 const float* __restrict__ W_ih,
                 const float* __restrict__ W_hh,
                 const float* __restrict__ b_ih,
                 const float* __restrict__ b_hh,
                 const float* __restrict__ embed_W,
                 const float* __restrict__ embed_b,
                 const float* __restrict__ pred_W,
                 const float* __restrict__ pred_b,
                 float* __restrict__ out, int N)
{
  extern __shared__ char smem[];
  _Float16* Wb    = (_Float16*)(smem + WB_OFF);
  _Float16* Acomb = (_Float16*)(smem + ACOMB_OFF);
  float* hf    = (float*)(smem + HF_OFF);
  float* pWh   = (float*)(smem + PWH_OFF);
  float* embW  = (float*)(smem + EMBW_OFF);
  float* embB  = (float*)(smem + EMBB_OFF);
  float* baseD = (float*)(smem + BASED_OFF);
  float* dispL = (float*)(smem + DISP_OFF);
  float* posL  = (float*)(smem + POS_OFF);
  float* red   = (float*)(smem + RED_OFF);

  const int tid  = threadIdx.x;
  const int lane = tid & 31, wave = tid >> 5;   // wave32
  const int lrow = lane & 15, half = lane >> 4;
  const int cg   = wave;                        // 4 waves -> 4 col-groups of 32
  const int nbase = blockIdx.x * MROWS;

  // ---- stage combined weights [W_ih^T ; W_hh^T] into LDS, K-major, f16 ----
  // Coalesced on the global side (consecutive lanes read consecutive k);
  // LDS takes the scatter (padded stride avoids systematic conflicts).
  for (int g = tid; g < GATES*EMBED; g += NTHREADS){
    int j = g >> 6, k = g & 63;
    Wb[k*WB_STRIDE + j] = (_Float16)W_ih[g];
  }
  for (int g = tid; g < GATES*HIDDEN; g += NTHREADS){
    int j = g >> 7, k = g & 127;
    Wb[(EMBED + k)*WB_STRIDE + j] = (_Float16)W_hh[g];
  }
  for (int i = tid; i < 256; i += NTHREADS)
    pWh[i] = pred_W[(i>>7)*(HIDDEN+CNN_OUT) + (i&127)];
  if (tid < 128) embW[tid] = embed_W[tid];
  if (tid < 64)  embB[tid] = embed_b[tid];

  // ---- per-lane persistent state ----
  float bias[8];
  #pragma unroll
  for (int g=0; g<4; ++g){
    #pragma unroll
    for (int s=0; s<2; ++s){
      int col = g*HIDDEN + cg*32 + s*16 + lrow;
      bias[g*2+s] = b_ih[col] + b_hh[col];
    }
  }
  int hist[8];
  #pragma unroll
  for (int v=0; v<8; ++v) hist[v] = obs_hist[nbase + v + 8*half];

  float h_reg[2][8], c_reg[2][8];
  #pragma unroll
  for (int s=0;s<2;++s){
    float hv = h0[cg*32 + s*16 + lrow];
    #pragma unroll
    for (int v=0; v<8; ++v){ h_reg[s][v]=hv; c_reg[s][v]=hv; }
  }
  store_h(Acomb, hf, h_reg, cg, lrow, half);
  __syncthreads();

  // ---- hoist ALL weight fragments into VGPRs (384 VGPRs, resident) ----
  v16h Ball[6][8];
  #pragma unroll
  for (int kc=0; kc<6; ++kc){
    #pragma unroll
    for (int g=0; g<4; ++g){
      #pragma unroll
      for (int s=0; s<2; ++s){
        Ball[kc][g*2+s] = load_frag_B(Wb, kc*32, g*HIDDEN + cg*32 + s*16, lane);
      }
    }
  }

  // ---- observation phase: 19 masked LSTM steps ----
  const int xrow = tid >> 3;            // row this thread embeds (0..15)
  const int xc0  = (tid & 7) * 8;       // its 8 embed columns
  for (int t = 1; t < OBS_LEN; ++t){
    float rx = obs_rel[((size_t)(nbase + xrow)*OBS_LEN + t)*2 + 0];
    float ry = obs_rel[((size_t)(nbase + xrow)*OBS_LEN + t)*2 + 1];
    write_xemb(Acomb, embW, embB, xrow, xc0, rx, ry);
    __syncthreads();                                  // A matrix complete
    lstm_step<true>(Acomb, Ball, bias, h_reg, c_reg, hist, OBS_LEN - t, lane);
    __syncthreads();                                  // all LDS reads done
    store_h(Acomb, hf, h_reg, cg, lrow, half);
  }

  // ---- step-invariant img @ pred_W[:,128:] + pred_b (compute once) ----
  {
    int row = tid >> 3, sub = tid & 7;
    const float4* ip = (const float4*)(img + (size_t)(nbase+row)*CNN_OUT + sub*256);
    const float4* w0 = (const float4*)(pred_W + HIDDEN + sub*256);
    const float4* w1 = (const float4*)(pred_W + (HIDDEN+CNN_OUT) + HIDDEN + sub*256);
    float a0=0.f, a1=0.f;
    #pragma unroll 4
    for (int q=0; q<64; ++q){
      float4 x = ip[q], u = w0[q], v = w1[q];
      a0 = fmaf(x.x,u.x, fmaf(x.y,u.y, fmaf(x.z,u.z, fmaf(x.w,u.w, a0))));
      a1 = fmaf(x.x,v.x, fmaf(x.y,v.y, fmaf(x.z,v.z, fmaf(x.w,v.w, a1))));
    }
    red[(row*8+sub)*2+0]=a0; red[(row*8+sub)*2+1]=a1;
  }
  __syncthreads();
  if (tid < 32){
    int row = tid >> 1, d = tid & 1;
    float s = pred_b[d];
    #pragma unroll
    for (int j=0;j<8;++j) s += red[(row*8+j)*2 + d];
    baseD[row*2+d] = s;
    posL[row*2+d]  = obs_pos[((size_t)(nbase+row)*OBS_LEN + (OBS_LEN-1))*2 + d];
  }

  // ---- prediction phase: 30 autoregressive steps ----
  for (int st = 0; st < PRED_LEN; ++st){
    __syncthreads();                  // h (f32 copy) and baseD/posL visible
    if (tid < 32){                    // disp = h . pW_h + (img part), f32 path
      int row = tid >> 1, d = tid & 1;
      float acc = baseD[row*2+d];
      const float* hp = hf + row*HF_STRIDE;
      const float* wp = pWh + d*HIDDEN;
      #pragma unroll 8
      for (int k=0;k<HIDDEN;++k) acc = fmaf(hp[k], wp[k], acc);
      float p = posL[row*2+d] + acc;
      posL[row*2+d] = p;
      out[(size_t)(nbase+row)*PRED_LEN*2 + st*2 + d] = p;
      dispL[row*2+d] = acc;
    }
    __syncthreads();                  // disp ready
    float rx = dispL[xrow*2+0], ry = dispL[xrow*2+1];
    write_xemb(Acomb, embW, embB, xrow, xc0, rx, ry);
    __syncthreads();                  // A matrix complete
    lstm_step<false>(Acomb, Ball, bias, h_reg, c_reg, hist, 0, lane);
    __syncthreads();                  // all LDS reads done
    store_h(Acomb, hf, h_reg, cg, lrow, half);
  }
}

extern "C" void kernel_launch(void* const* d_in, const int* in_sizes, int n_in,
                              void* d_out, int out_size, void* d_ws, size_t ws_size,
                              hipStream_t stream) {
  (void)n_in; (void)out_size; (void)d_ws; (void)ws_size;
  const float* img   = (const float*)d_in[0];
  const float* opos  = (const float*)d_in[1];
  const int*   hist  = (const int*)  d_in[2];
  const float* orel  = (const float*)d_in[3];
  const float* h0    = (const float*)d_in[4];
  const float* W_ih  = (const float*)d_in[5];
  const float* W_hh  = (const float*)d_in[6];
  const float* b_ih  = (const float*)d_in[7];
  const float* b_hh  = (const float*)d_in[8];
  const float* embW  = (const float*)d_in[9];
  const float* embB  = (const float*)d_in[10];
  const float* predW = (const float*)d_in[11];
  const float* predB = (const float*)d_in[12];
  const int N = in_sizes[2];                 // obs_hist_size has N elements

  dim3 grid(N / MROWS), block(NTHREADS);
  // SMEM_BYTES (~212.6 KB) targets the gfx1250 WGP's 320 KB LDS (ISA §3.1).
  hipLaunchKernelGGL(social_lstm_wmma, grid, block, SMEM_BYTES, stream,
                     img, opos, hist, orel, h0, W_ih, W_hh, b_ih, b_hh,
                     embW, embB, predW, predB, (float*)d_out, N);
}
